// CRF_13091060318391
// MI455X (gfx1250) — compile-verified, matrix operand order checked by reading, not run on hardware
//
#include <hip/hip_runtime.h>

typedef __attribute__((ext_vector_type(16))) _Float16     v16h;
typedef __attribute__((ext_vector_type(8)))  float        v8f;
typedef __attribute__((ext_vector_type(4)))  unsigned int v4u;
typedef __attribute__((ext_vector_type(8)))  int          v8i;
typedef __attribute__((ext_vector_type(4)))  int          v4i;

#define Bsz 128
#define Tsz 2048
#define Ksz 96
#define EP  104   // padded f16 row stride (halves)
#define AP  100   // padded f32 row stride (floats)

// ---------------------------------------------------------------------------
// TDM: DMA one emission tile (16 rows x 96 f32, row stride T*K) into LDS.
// D# per CDNA5 ISA section 8 (groups 0/1; groups 2/3 zero => 2D tensor).
// ---------------------------------------------------------------------------
__device__ __forceinline__ void tdm_load_tile(unsigned lds_off, const float* gptr) {
    unsigned long long ga = (unsigned long long)(uintptr_t)gptr;
    v4u g0;
    g0.x = 1u;                                                 // count=1 (valid), no gather
    g0.y = lds_off;                                            // lds_addr (bytes)
    g0.z = (unsigned)(ga & 0xFFFFFFFFu);                       // global_addr[31:0]
    g0.w = (unsigned)((ga >> 32) & 0x01FFFFFFu) | (2u << 30);  // global_addr[56:32] | type=2
    v8i g1;
    g1[0] = (int)(2u << 16);                   // workgroup_mask=0, data_size=2 (4 bytes)
    g1[1] = (int)((unsigned)Ksz << 16);        // tensor_dim0 = 96 (bits 63:48)
    g1[2] = (int)(16u << 16);                  // tensor_dim1 = 16 (bits 95:80)
    g1[3] = (int)((unsigned)Ksz << 16);        // tile_dim0 = 96 (bits 127:112)
    g1[4] = 16;                                // tile_dim1 = 16 (bits 143:128), tile_dim2=0
    g1[5] = (int)(Tsz * Ksz);                  // tensor_dim0_stride = T*K (bits 191:160)
    g1[6] = 0;
    g1[7] = 0;
    v4i gz = {0, 0, 0, 0};
#if __has_include(<hip/amd_detail/amd_gfx1250_TDM.h>)
    v8i gz8 = {0, 0, 0, 0, 0, 0, 0, 0};
    __builtin_amdgcn_tensor_load_to_lds(g0, g1, gz, gz, gz8, 0);   // clang-23 / therock arity
#else
    __builtin_amdgcn_tensor_load_to_lds(g0, g1, gz, gz, 0);        // ROCm 7.2 clang-22 arity
#endif
}

// ---------------------------------------------------------------------------
// Kernel 1: forward algorithm (log partition) via WMMA in the exp-domain.
// One block = one 16-row batch tile; 6 waves, wave w owns output cols [16w,16w+16).
// alpha'[b,j] = mask ? ( log( (exp(alpha-m) @ exp(T))[b,j] ) + m[b] + emit[b,j] ) : alpha[b,j]
// 3 barriers per step: alpha double-buffered, row-max redundantly reduced per
// thread, emissions stream in one step ahead via TENSOR_LOAD_TO_LDS.
// ---------------------------------------------------------------------------
__global__ __launch_bounds__(192)
void crf_forward_denominator(const float* __restrict__ inp,
                             const int*   __restrict__ maskp,
                             const float* __restrict__ trans,
                             float*       __restrict__ ws_den)
{
    __shared__ _Float16 E[Ksz][EP];                  // exp(transitions), K-major
    __shared__ _Float16 P[16][EP];                   // exp(alpha - rowmax), f16
    __shared__ float    A[2][16][AP];                // alpha, double-buffered
    __shared__ float    part[16][12];                // strip partials (16B-aligned rows)
    __shared__ float    mrow[16];                    // per-row max
    __shared__ int      mskrow[16];                  // per-row mask at current t
    __shared__ __align__(16) float Ebuf[2][16][Ksz]; // TDM emission double buffer

    const int tid   = threadIdx.x;
    const int lane  = tid & 31;
    const int w     = tid >> 5;        // wave id 0..5 -> N-block
    const int bt    = blockIdx.x;      // batch tile 0..7
    const int laneN = lane & 15;
    const int hi    = (lane >> 4) & 1; // lane half (ISA fragment tables)

    // Precompute E = exp(transitions) once (reused 2047 times).
    for (int idx = tid; idx < Ksz * Ksz; idx += 192) {
        int i = idx / Ksz, j = idx % Ksz;
        E[i][j] = (_Float16)__expf(trans[idx]);
    }
    // alpha0 = inputs[:, 0, :] -> buffer 0
    for (int idx = tid; idx < 16 * Ksz; idx += 192) {
        int r = idx / Ksz, c = idx % Ksz;
        A[0][r][c] = inp[((size_t)(bt * 16 + r) * Tsz) * Ksz + c];
    }
    __syncthreads();

    // Preload B fragments for this wave's N-block: constant across all timesteps.
    // 16-bit operand layout (ISA 7.12.2): element e -> VGPR e/2, pos e%2;
    // k = lane_half*8 + (v%4)*2 + pos + (v/4)*16.
    v16h Bf[3];
    #pragma unroll
    for (int kt = 0; kt < 3; ++kt) {
        #pragma unroll
        for (int e = 0; e < 16; ++e) {
            int k = kt * 32 + hi * 8 + ((e >> 1) & 3) * 2 + (e & 1) + (e >> 3) * 16;
            Bf[kt][e] = E[k][w * 16 + laneN];
        }
    }

    const unsigned ebuf_lds0 = (unsigned)(uintptr_t)(&Ebuf[0][0][0]);
    const unsigned ebuf_lds1 = (unsigned)(uintptr_t)(&Ebuf[1][0][0]);

    // Kick off TDM for the first timestep's emission tile (t=1 -> buffer 0).
    if (tid == 0)
        tdm_load_tile(ebuf_lds0, inp + ((size_t)(bt * 16) * Tsz + 1) * Ksz);

    // Prefetch mask words for t=1 into registers (threads 0..15 own one row each).
    int mreg = 1;
    if (tid < 16) mreg = maskp[(size_t)(bt * 16 + tid) * Tsz + 1];

    const int rrow  = tid / 12;        // strip-reduction row (192 = 16 rows x 12 strips)
    const int cbase = (tid % 12) * 8;  // 8 cols per strip

    for (int t = 1; t < Tsz; ++t) {
        const int cur = (t - 1) & 1;   // alpha read buffer
        const int nxt = t & 1;         // alpha write buffer
        const int buf = (t - 1) & 1;   // emission buffer for this step

        // --- phase A: strip maxima of alpha --------------------------------
        float mx = A[cur][rrow][cbase];
        #pragma unroll
        for (int c = 1; c < 8; ++c) mx = fmaxf(mx, A[cur][rrow][cbase + c]);
        part[rrow][tid % 12] = mx;
        __syncthreads();                                       // B1

        // --- phase B: every thread folds its row's partials (no 2nd barrier),
        //     P = exp(alpha - m) as f16; tid<16 publishes mrow/mask; wave 0
        //     launches next DMA and fences the current tile.
        float m = part[rrow][0];
        #pragma unroll
        for (int c = 1; c < 12; ++c) m = fmaxf(m, part[rrow][c]);
        #pragma unroll
        for (int c = 0; c < 8; ++c)
            P[rrow][cbase + c] = (_Float16)__expf(A[cur][rrow][cbase + c] - m);
        if (tid < 16) {
            float mm = part[tid][0];
            #pragma unroll
            for (int c = 1; c < 12; ++c) mm = fmaxf(mm, part[tid][c]);
            mrow[tid]   = mm;
            mskrow[tid] = mreg;
        }
        if (tid == 0) {
            if (t + 1 < Tsz) {
                tdm_load_tile(buf ? ebuf_lds0 : ebuf_lds1,
                              inp + ((size_t)(bt * 16) * Tsz + (t + 1)) * Ksz);
                __builtin_amdgcn_s_wait_tensorcnt(1);  // all but newest done -> Ebuf[buf] ready
            } else {
                __builtin_amdgcn_s_wait_tensorcnt(0);
            }
        }
        __syncthreads();                                       // B2

        // --- phase C: acc = P @ E via 3x v_wmma_f32_16x16x32_f16 -----------
        v16h Af[3];
        #pragma unroll
        for (int kt = 0; kt < 3; ++kt) {
            #pragma unroll
            for (int e = 0; e < 16; ++e) {
                int k = kt * 32 + hi * 8 + ((e >> 1) & 3) * 2 + (e & 1) + (e >> 3) * 16;
                Af[kt][e] = P[laneN][k];
            }
        }
        v8f acc = {};
        acc = __builtin_amdgcn_wmma_f32_16x16x32_f16(false, Af[0], false, Bf[0], (short)0, acc, false, false);
        acc = __builtin_amdgcn_wmma_f32_16x16x32_f16(false, Af[1], false, Bf[1], (short)0, acc, false, false);
        acc = __builtin_amdgcn_wmma_f32_16x16x32_f16(false, Af[2], false, Bf[2], (short)0, acc, false, false);

        // C/D layout: VGPR r -> row M = r + 8*hi, col N = laneN (+ 16w globally).
        // Write goes to the other alpha buffer -> no pre-write barrier needed.
        #pragma unroll
        for (int r = 0; r < 8; ++r) {
            int   M    = r + hi * 8;
            float emit = Ebuf[buf][M][w * 16 + laneN];
            float oldv = A[cur][M][w * 16 + laneN];
            float nv   = __logf(acc[r]) + mrow[M] + emit;
            A[nxt][M][w * 16 + laneN] = (mskrow[M] > 0) ? nv : oldv;
        }
        // Prefetch next step's mask words while stores drain.
        if (tid < 16 && t + 1 < Tsz)
            mreg = maskp[(size_t)(bt * 16 + tid) * Tsz + (t + 1)];
        __syncthreads();                                       // B3
    }

    // --- final: den[b] = logsumexp_j alpha[b][j], alpha in buffer (Tsz-1)&1 --
    const int fin = (Tsz - 1) & 1;
    float mx = A[fin][rrow][cbase];
    #pragma unroll
    for (int c = 1; c < 8; ++c) mx = fmaxf(mx, A[fin][rrow][cbase + c]);
    part[rrow][tid % 12] = mx;
    __syncthreads();
    float m = part[rrow][0];
    #pragma unroll
    for (int c = 1; c < 12; ++c) m = fmaxf(m, part[rrow][c]);
    float s = 0.f;
    #pragma unroll
    for (int c = 0; c < 8; ++c) s += __expf(A[fin][rrow][cbase + c] - m);
    part[rrow][tid % 12] = s;   // safe: each thread rewrites its own slot
    __syncthreads();
    if (tid < 16) {
        float mm = part[tid][0];   // NOTE: part now holds sums; recompute max first
        // recompute row max from alpha directly (cheap, once)
        float rm = A[fin][tid][0];
        #pragma unroll
        for (int c = 1; c < Ksz; ++c) rm = fmaxf(rm, A[fin][tid][c]);
        float ss = 0.f;
        #pragma unroll
        for (int c = 0; c < 12; ++c) ss += part[tid][c];
        (void)mm;
        ws_den[bt * 16 + tid] = rm + __logf(ss);
    }
}

// ---------------------------------------------------------------------------
// Kernel 2: gold-path score (numerator). One block per batch row,
// deterministic LDS tree reduction.
// ---------------------------------------------------------------------------
__global__ __launch_bounds__(256)
void crf_numerator(const float* __restrict__ inp,
                   const int*   __restrict__ tags,
                   const int*   __restrict__ maskp,
                   const float* __restrict__ trans,
                   float*       __restrict__ ws_num)
{
    __shared__ float sred[256];
    __shared__ int   cred[256];
    const int b   = blockIdx.x;
    const int tid = threadIdx.x;
    const int*   tg  = tags  + (size_t)b * Tsz;
    const int*   mk  = maskp + (size_t)b * Tsz;
    const float* ein = inp   + (size_t)b * Tsz * Ksz;

    float s = 0.f; int cnt = 0;
    for (int t = tid; t < Tsz; t += 256) {
        cnt += (mk[t] > 0) ? 1 : 0;
        if (t < Tsz - 1) {
            s += trans[tg[t] * Ksz + tg[t + 1]] * (float)mk[t + 1]; // transition, gated by mask[t+1]
            s += ein[(size_t)t * Ksz + tg[t]] * (float)mk[t];       // emission,  gated by mask[t]
        }
    }
    sred[tid] = s; cred[tid] = cnt;
    __syncthreads();
    for (int off = 128; off > 0; off >>= 1) {
        if (tid < off) { sred[tid] += sred[tid + off]; cred[tid] += cred[tid + off]; }
        __syncthreads();
    }
    if (tid == 0) {
        int last_idx = cred[0] - 1;
        if (last_idx < 0) last_idx = 0;
        int last_tag = tg[last_idx];
        ws_num[b] = sred[0] + ein[(size_t)(Tsz - 1) * Ksz + last_tag] * (float)mk[Tsz - 1];
    }
}

// ---------------------------------------------------------------------------
// Kernel 3: deterministic serial reduction (no float atomics).
// ---------------------------------------------------------------------------
__global__ void crf_final(const float* __restrict__ ws_den,
                          const float* __restrict__ ws_num,
                          float* __restrict__ out)
{
    if (threadIdx.x == 0 && blockIdx.x == 0) {
        float acc = 0.f;
        for (int b = 0; b < Bsz; ++b) acc += ws_num[b] - ws_den[b];
        out[0] = acc;
    }
}

extern "C" void kernel_launch(void* const* d_in, const int* in_sizes, int n_in,
                              void* d_out, int out_size, void* d_ws, size_t ws_size,
                              hipStream_t stream) {
    const float* inp   = (const float*)d_in[0];   // [B,T,K] f32
    const int*   tags  = (const int*)  d_in[1];   // [B,T] i32
    const int*   mask  = (const int*)  d_in[2];   // [B,T] i32
    const float* trans = (const float*)d_in[3];   // [K,K] f32
    float* ws_den = (float*)d_ws;                 // [B]
    float* ws_num = ws_den + Bsz;                 // [B]

    crf_forward_denominator<<<dim3(Bsz / 16), dim3(192), 0, stream>>>(inp, mask, trans, ws_den);
    crf_numerator<<<dim3(Bsz), dim3(256), 0, stream>>>(inp, tags, mask, trans, ws_num);
    crf_final<<<1, 1, 0, stream>>>(ws_den, ws_num, (float*)d_out);
}
